// HeposMultiHeadAttention_24730421690606
// MI455X (gfx1250) — matrix operand order repaired and checked by reading, not compiled, not run on hardware
//
#include <hip/hip_runtime.h>

// ---------------- problem constants ----------------
#define BATCH   2
#define TGTLEN  1024
#define SRCLEN  8192
#define DMODEL  1024
#define NHEAD   8
#define HDIM    128
#define HSTRIDE 8
#define TPH     1024   // keys per head after HEPOS packing = SRCLEN / HSTRIDE

typedef __bf16 bf16;
typedef __bf16 bf16x8  __attribute__((ext_vector_type(8)));
typedef __bf16 bf16x16 __attribute__((ext_vector_type(16)));
typedef float  f32x8   __attribute__((ext_vector_type(8)));

static __device__ __forceinline__ f32x8 zero8() {
  f32x8 z;
#pragma unroll
  for (int i = 0; i < 8; ++i) z[i] = 0.0f;
  return z;
}

// A fragment (16x32 MxK bf16) built from an fp32 row-major source.
// Layout per CDNA5 ISA 7.12.2: lane m=lane&15, half h=lane>>4;
// VGPR0..3 hold K = 8h+0..7, VGPR4..7 hold K = 16+8h+0..7 (2 bf16 per VGPR).
static __device__ __forceinline__ bf16x16 a_frag_f32(const float* base, int lda, int lane) {
  const int m = lane & 15, h = lane >> 4;
  const float* p = base + (size_t)m * lda + 8 * h;
  bf16x16 f;
#pragma unroll
  for (int i = 0; i < 8; ++i) {
    f[i]     = (bf16)p[i];
    f[i + 8] = (bf16)p[16 + i];
  }
  return f;
}

// Same A fragment from a bf16 row-major source (two 16B loads per lane).
static __device__ __forceinline__ bf16x16 a_frag_bf16(const bf16* base, int lda, int lane) {
  const int m = lane & 15, h = lane >> 4;
  const bf16* p = base + (size_t)m * lda + 8 * h;
  bf16x8 lo = *(const bf16x8*)p;
  bf16x8 hi = *(const bf16x8*)(p + 16);
  bf16x16 f;
#pragma unroll
  for (int i = 0; i < 8; ++i) { f[i] = lo[i]; f[i + 8] = hi[i]; }
  return f;
}

// B fragment (32x16 KxN bf16). Memory holds B column-major (i.e. rows are the N
// columns, contiguous in K). Per ISA: lane n=lane&15 holds column n; half h
// carries K = 8h+0..7 then K = 16+8h+0..7 (2 per VGPR) -> two 16-byte runs
// per lane.
static __device__ __forceinline__ bf16x16 b_frag(const bf16* base, int ldk, int lane) {
  const int n = lane & 15, h = lane >> 4;
  const bf16* p = base + (size_t)n * ldk + 8 * h;
  bf16x8 lo = *(const bf16x8*)p;
  bf16x8 hi = *(const bf16x8*)(p + 16);
  bf16x16 f;
#pragma unroll
  for (int i = 0; i < 8; ++i) { f[i] = lo[i]; f[i + 8] = hi[i]; }
  return f;
}

static __device__ __forceinline__ f32x8 wmma_bf16(bf16x16 a, bf16x16 b, f32x8 c) {
  // (neg_a, A, neg_b, B, c_mod, C, reuse_a, reuse_b)
  return __builtin_amdgcn_wmma_f32_16x16x32_bf16(false, a, false, b, (short)0, c,
                                                 false, false);
}

// ---------------- kernel 1: weight convert + transpose ----------------
// W is [d_in=1024][d_out=1024] fp32; produce WT bf16 [d_out][d_in] so that
// b_frag() reads are contiguous per lane.
__global__ void wt_cvt_kernel(const float* __restrict__ W, bf16* __restrict__ WT) {
  const int idx = blockIdx.x * blockDim.x + threadIdx.x;   // over 1024*1024
  const int n = idx >> 10;
  const int k = idx & 1023;
  WT[idx] = (bf16)W[(size_t)k * DMODEL + n];
}

// ---------------- kernel 2: Q projection ----------------
// out(bf16)[row][col] = sum_k A(f32)[row][k] * W[k][col] + bias[col]
// block = 256 thr = 8 waves (2x4), wave tile 32x32 (2x2 WMMA), block tile 64x128.
__global__ void __launch_bounds__(256)
qproj_kernel(const float* __restrict__ A, const bf16* __restrict__ WT,
             const float* __restrict__ bias, bf16* __restrict__ out) {
  const int lane = threadIdx.x & 31, wid = threadIdx.x >> 5;
  const int wm = wid >> 2, wn = wid & 3;
  const int r0 = blockIdx.x * 64 + wm * 32;
  const int c0 = blockIdx.y * 128 + wn * 32;

  f32x8 acc[2][2];
  acc[0][0] = zero8(); acc[0][1] = zero8();
  acc[1][0] = zero8(); acc[1][1] = zero8();

  for (int kk = 0; kk < DMODEL; kk += 32) {
    bf16x16 a0 = a_frag_f32(A + (size_t)r0 * DMODEL + kk, DMODEL, lane);
    bf16x16 a1 = a_frag_f32(A + (size_t)(r0 + 16) * DMODEL + kk, DMODEL, lane);
    bf16x16 b0 = b_frag(WT + (size_t)c0 * DMODEL + kk, DMODEL, lane);
    bf16x16 b1 = b_frag(WT + (size_t)(c0 + 16) * DMODEL + kk, DMODEL, lane);
    acc[0][0] = wmma_bf16(a0, b0, acc[0][0]);
    acc[0][1] = wmma_bf16(a0, b1, acc[0][1]);
    acc[1][0] = wmma_bf16(a1, b0, acc[1][0]);
    acc[1][1] = wmma_bf16(a1, b1, acc[1][1]);
  }

  const int n = lane & 15, hh = lane >> 4;
#pragma unroll
  for (int i = 0; i < 2; ++i)
#pragma unroll
    for (int j = 0; j < 2; ++j)
#pragma unroll
      for (int r = 0; r < 8; ++r) {
        const int row = r0 + i * 16 + r + 8 * hh;
        const int col = c0 + j * 16 + n;
        out[(size_t)row * DMODEL + col] = (bf16)(acc[i][j][r] + bias[col]);
      }
}

// ---------------- kernel 3: packed K/V projection ----------------
// For (b,h): rows j = h + 8t (t=0..1023) of the source, and ONLY output columns
// h*128..h*128+127 (HEPOS: head h never reads anything else). The stride-8 row
// gather is just lda = 8*DMODEL. transposeOut=0 -> Kp[t][d]; =1 -> VpT[d][t].
__global__ void __launch_bounds__(256)
kvproj_kernel(const float* __restrict__ X, const bf16* __restrict__ WT,
              const float* __restrict__ bias, bf16* __restrict__ outp,
              int transposeOut) {
  const int bh = blockIdx.y;
  const int b = bh >> 3, h = bh & 7;
  const int lane = threadIdx.x & 31, wid = threadIdx.x >> 5;
  const int wm = wid >> 2, wn = wid & 3;
  const int r0 = blockIdx.x * 64 + wm * 32;   // t rows
  const int c0 = wn * 32;                     // head-local dim (N = 128)
  const int lda = HSTRIDE * DMODEL;           // 8192 floats between packed rows

  const float* Ab   = X + (size_t)b * SRCLEN * DMODEL + (size_t)h * DMODEL;
  const bf16*  WTb  = WT + (size_t)(h * HDIM) * DMODEL;
  const float* bb   = bias + h * HDIM;
  bf16*        ob   = outp + (size_t)bh * (TPH * HDIM);

  f32x8 acc[2][2];
  acc[0][0] = zero8(); acc[0][1] = zero8();
  acc[1][0] = zero8(); acc[1][1] = zero8();

  for (int kk = 0; kk < DMODEL; kk += 32) {
    bf16x16 a0 = a_frag_f32(Ab + (size_t)r0 * lda + kk, lda, lane);
    bf16x16 a1 = a_frag_f32(Ab + (size_t)(r0 + 16) * lda + kk, lda, lane);
    bf16x16 b0 = b_frag(WTb + (size_t)c0 * DMODEL + kk, DMODEL, lane);
    bf16x16 b1 = b_frag(WTb + (size_t)(c0 + 16) * DMODEL + kk, DMODEL, lane);
    acc[0][0] = wmma_bf16(a0, b0, acc[0][0]);
    acc[0][1] = wmma_bf16(a0, b1, acc[0][1]);
    acc[1][0] = wmma_bf16(a1, b0, acc[1][0]);
    acc[1][1] = wmma_bf16(a1, b1, acc[1][1]);
  }

  const int n = lane & 15, hh = lane >> 4;
#pragma unroll
  for (int i = 0; i < 2; ++i)
#pragma unroll
    for (int j = 0; j < 2; ++j)
#pragma unroll
      for (int r = 0; r < 8; ++r) {
        const int row = r0 + i * 16 + r + 8 * hh;   // t
        const int col = c0 + j * 16 + n;            // d
        const float val = acc[i][j][r] + bb[col];
        if (transposeOut) ob[(size_t)col * TPH + row] = (bf16)val;
        else              ob[(size_t)row * HDIM + col] = (bf16)val;
      }
}

// ---------------- kernel 4: fused attention per (b, h, 32-query tile) ----------
// 320KB LDS/WGP lets the whole 32x1024 fp32 score slab live in LDS:
//   phase 1: S = (Q Kp^T) * 1/sqrt(128)  (all keys are legal after packing)
//   phase 2: exact softmax over the 1024 packed keys (exp left in S, 1/sum saved)
//   phase 3: O = P V, scaled by 1/sum at store
#define ATTN_TQ 32
#define ATTN_LD 1032   // padded row stride (floats) to dodge LDS bank conflicts
#define ATTN_SMEM_FLOATS (ATTN_TQ * ATTN_LD + ATTN_TQ * 8 + ATTN_TQ)

__global__ void __launch_bounds__(256)
attn_kernel(const bf16* __restrict__ Qp, const bf16* __restrict__ Kp,
            const bf16* __restrict__ VpT, bf16* __restrict__ ctx) {
  extern __shared__ float smem[];
  float* S       = smem;                       // [ATTN_TQ][ATTN_LD]
  float* red     = smem + ATTN_TQ * ATTN_LD;   // [ATTN_TQ][8]
  float* rowstat = red + ATTN_TQ * 8;          // [ATTN_TQ]

  const int qt0 = blockIdx.x * ATTN_TQ;
  const int h = blockIdx.y, b = blockIdx.z;
  const int tid = threadIdx.x, lane = tid & 31, wid = tid >> 5;

  const bf16* Qb = Qp  + ((size_t)(b * TGTLEN + qt0)) * DMODEL + h * HDIM;
  const bf16* Kb = Kp  + (size_t)(b * NHEAD + h) * ((size_t)TPH * HDIM);
  const bf16* Vb = VpT + (size_t)(b * NHEAD + h) * ((size_t)HDIM * TPH);
  bf16*       Cb = ctx + ((size_t)(b * TGTLEN + qt0)) * DMODEL + h * HDIM;

  // ---- phase 1: scores ----
  {
    const int wm = wid >> 2;   // 16-row group (2 of them)
    const int wn = wid & 3;    // 256-key slab (4 of them)
    bf16x16 qf[4];
#pragma unroll
    for (int kq = 0; kq < 4; ++kq)
      qf[kq] = a_frag_bf16(Qb + (size_t)(wm * 16) * DMODEL + kq * 32, DMODEL, lane);
    const float scale = 0.088388347648318447f;  // 1/sqrt(HDIM)
    const int nn = lane & 15, hh = lane >> 4;
    for (int cn = wn * 256; cn < wn * 256 + 256; cn += 16) {
      f32x8 acc = zero8();
#pragma unroll
      for (int kq = 0; kq < 4; ++kq) {
        bf16x16 kf = b_frag(Kb + (size_t)cn * HDIM + kq * 32, HDIM, lane);
        acc = wmma_bf16(qf[kq], kf, acc);
      }
#pragma unroll
      for (int r = 0; r < 8; ++r) {
        const int row = wm * 16 + r + 8 * hh;
        S[row * ATTN_LD + cn + nn] = acc[r] * scale;
      }
    }
  }
  __syncthreads();

  // ---- phase 2: softmax (8 threads per row) ----
  {
    const int row = tid >> 3, sub = tid & 7;
    float mx = -1e30f;
    for (int c = sub; c < TPH; c += 8) mx = fmaxf(mx, S[row * ATTN_LD + c]);
    red[row * 8 + sub] = mx;
    __syncthreads();
    if (sub == 0) {
      float m2 = red[row * 8];
#pragma unroll
      for (int i = 1; i < 8; ++i) m2 = fmaxf(m2, red[row * 8 + i]);
      rowstat[row] = m2;
    }
    __syncthreads();
    const float rmax = rowstat[row];
    float sum = 0.0f;
    for (int c = sub; c < TPH; c += 8) {
      const float e = __expf(S[row * ATTN_LD + c] - rmax);
      S[row * ATTN_LD + c] = e;
      sum += e;
    }
    red[row * 8 + sub] = sum;
    __syncthreads();
    if (sub == 0) {
      float s2 = 0.0f;
#pragma unroll
      for (int i = 0; i < 8; ++i) s2 += red[row * 8 + i];
      rowstat[row] = 1.0f / s2;
    }
  }
  __syncthreads();

  // ---- phase 3: O = P V ----
  {
    const int wm = wid >> 2;   // 16-row group
    const int wn = wid & 3;    // 32-col group of HDIM
    f32x8 acc[2];
    acc[0] = zero8(); acc[1] = zero8();
    for (int kk = 0; kk < TPH; kk += 32) {
      bf16x16 af = a_frag_f32(S + (size_t)(wm * 16) * ATTN_LD + kk, ATTN_LD, lane);
#pragma unroll
      for (int j = 0; j < 2; ++j) {
        bf16x16 vf = b_frag(Vb + (size_t)(wn * 32 + j * 16) * TPH + kk, TPH, lane);
        acc[j] = wmma_bf16(af, vf, acc[j]);
      }
    }
    const int nn = lane & 15, hh = lane >> 4;
#pragma unroll
    for (int j = 0; j < 2; ++j)
#pragma unroll
      for (int r = 0; r < 8; ++r) {
        const int row = wm * 16 + r + 8 * hh;
        const float v = acc[j][r] * rowstat[row];
        Cb[(size_t)row * DMODEL + wn * 32 + j * 16 + nn] = (bf16)v;
      }
  }
}

// ---------------- kernel 5: output projection (bf16 A, fp32 out) -------------
__global__ void __launch_bounds__(256)
oproj_kernel(const bf16* __restrict__ A, const bf16* __restrict__ WT,
             const float* __restrict__ bias, float* __restrict__ out) {
  const int lane = threadIdx.x & 31, wid = threadIdx.x >> 5;
  const int wm = wid >> 2, wn = wid & 3;
  const int r0 = blockIdx.x * 64 + wm * 32;
  const int c0 = blockIdx.y * 128 + wn * 32;

  f32x8 acc[2][2];
  acc[0][0] = zero8(); acc[0][1] = zero8();
  acc[1][0] = zero8(); acc[1][1] = zero8();

  for (int kk = 0; kk < DMODEL; kk += 32) {
    bf16x16 a0 = a_frag_bf16(A + (size_t)r0 * DMODEL + kk, DMODEL, lane);
    bf16x16 a1 = a_frag_bf16(A + (size_t)(r0 + 16) * DMODEL + kk, DMODEL, lane);
    bf16x16 b0 = b_frag(WT + (size_t)c0 * DMODEL + kk, DMODEL, lane);
    bf16x16 b1 = b_frag(WT + (size_t)(c0 + 16) * DMODEL + kk, DMODEL, lane);
    acc[0][0] = wmma_bf16(a0, b0, acc[0][0]);
    acc[0][1] = wmma_bf16(a0, b1, acc[0][1]);
    acc[1][0] = wmma_bf16(a1, b0, acc[1][0]);
    acc[1][1] = wmma_bf16(a1, b1, acc[1][1]);
  }

  const int n = lane & 15, hh = lane >> 4;
#pragma unroll
  for (int i = 0; i < 2; ++i)
#pragma unroll
    for (int j = 0; j < 2; ++j)
#pragma unroll
      for (int r = 0; r < 8; ++r) {
        const int row = r0 + i * 16 + r + 8 * hh;
        const int col = c0 + j * 16 + n;
        out[(size_t)row * DMODEL + col] = acc[i][j][r] + bias[col];
      }
}

// ---------------- launch ----------------
extern "C" void kernel_launch(void* const* d_in, const int* in_sizes, int n_in,
                              void* d_out, int out_size, void* d_ws, size_t ws_size,
                              hipStream_t stream) {
  (void)in_sizes; (void)n_in; (void)out_size; (void)ws_size;
  const float* q  = (const float*)d_in[0];
  const float* k  = (const float*)d_in[1];
  const float* v  = (const float*)d_in[2];
  const float* Wq = (const float*)d_in[3];
  const float* bq = (const float*)d_in[4];
  const float* Wk = (const float*)d_in[5];
  const float* bk = (const float*)d_in[6];
  const float* Wv = (const float*)d_in[7];
  const float* bv = (const float*)d_in[8];
  const float* Wo = (const float*)d_in[9];
  const float* bo = (const float*)d_in[10];
  float* out = (float*)d_out;

  // workspace map (24 MB total):
  char* ws = (char*)d_ws;
  const size_t MB = 1024 * 1024;
  bf16* WTq = (bf16*)(ws + 0 * MB);   // 2 MB each: bf16 W^T
  bf16* WTk = (bf16*)(ws + 2 * MB);
  bf16* WTv = (bf16*)(ws + 4 * MB);
  bf16* WTo = (bf16*)(ws + 6 * MB);
  bf16* Qp  = (bf16*)(ws + 8 * MB);   // [B*TGT][D] bf16, 4 MB
  bf16* Kp  = (bf16*)(ws + 12 * MB);  // [B][H][TPH][HDIM] bf16, 4 MB
  bf16* VpT = (bf16*)(ws + 16 * MB);  // [B][H][HDIM][TPH] bf16, 4 MB
  bf16* Ctx = (bf16*)(ws + 20 * MB);  // [B*TGT][D] bf16, 4 MB

  // 1) weights -> bf16, transposed
  wt_cvt_kernel<<<dim3(4096), dim3(256), 0, stream>>>(Wq, WTq);
  wt_cvt_kernel<<<dim3(4096), dim3(256), 0, stream>>>(Wk, WTk);
  wt_cvt_kernel<<<dim3(4096), dim3(256), 0, stream>>>(Wv, WTv);
  wt_cvt_kernel<<<dim3(4096), dim3(256), 0, stream>>>(Wo, WTo);

  // 2) Q projection: [2048 x 1024] x [1024 x 1024]
  qproj_kernel<<<dim3((BATCH * TGTLEN) / 64, DMODEL / 128), dim3(256), 0, stream>>>(
      q, WTq, bq, Qp);

  // 3) K/V packed projections (only the owning head's 128 dims per row)
  kvproj_kernel<<<dim3(TPH / 64, BATCH * NHEAD), dim3(256), 0, stream>>>(
      k, WTk, bk, Kp, 0);
  kvproj_kernel<<<dim3(TPH / 64, BATCH * NHEAD), dim3(256), 0, stream>>>(
      v, WTv, bv, VpT, 1);

  // 4) attention (dynamic LDS: 133 KB, within CDNA5's 320 KB/WGP)
  const size_t smem = ATTN_SMEM_FLOATS * sizeof(float);
  hipFuncSetAttribute(reinterpret_cast<const void*>(attn_kernel),
                      hipFuncAttributeMaxDynamicSharedMemorySize, (int)smem);
  attn_kernel<<<dim3(TGTLEN / ATTN_TQ, NHEAD, BATCH), dim3(256), smem, stream>>>(
      Qp, Kp, VpT, Ctx);

  // 5) output projection -> fp32 result
  oproj_kernel<<<dim3((BATCH * TGTLEN) / 64, DMODEL / 128), dim3(256), 0, stream>>>(
      Ctx, WTo, bo, out);
}
